// Sparse_3d_Residual_Block_69526930588488
// MI455X (gfx1250) — matrix operand order repaired and checked by reading, not compiled, not run on hardware
//
#include <hip/hip_runtime.h>

typedef __attribute__((ext_vector_type(16))) __bf16         v16bf;
typedef __attribute__((ext_vector_type(8)))  float          v8f;
typedef __attribute__((ext_vector_type(8)))  unsigned short u16x8;
typedef __attribute__((ext_vector_type(16))) unsigned short u16x16;
typedef __attribute__((ext_vector_type(4)))  int            v4i;

#define CIN       96
#define COUT      96
#define KOFF      27
#define M_TILE    128
#define SA_STRIDE 104   // bf16 elems per LDS row (padded vs 64-bank conflicts)
#define SB_STRIDE 104
#define EPSV      1e-5f

// ---------------- bf16 helpers ----------------
__device__ __forceinline__ unsigned short f2bf(float f) {
  unsigned int u = __float_as_uint(f);
  u += 0x7FFFu + ((u >> 16) & 1u);      // round-to-nearest-even
  return (unsigned short)(u >> 16);
}
__device__ __forceinline__ unsigned long long pack4bf(float x, float y, float z, float w) {
  unsigned long long lo = (unsigned int)f2bf(x) | ((unsigned int)f2bf(y) << 16);
  unsigned long long hi = (unsigned int)f2bf(z) | ((unsigned int)f2bf(w) << 16);
  return lo | (hi << 32);
}

// ---------------- async global->LDS copy (CDNA5 DMA path) ----------------
#if __has_builtin(__builtin_amdgcn_global_load_async_to_lds_b128)
#define HAS_ASYNC_LDS 1
#else
#define HAS_ASYNC_LDS 0
#endif

__device__ __forceinline__ void copy_chunk_b128(const unsigned short* g, unsigned short* l) {
#if HAS_ASYNC_LDS
  __builtin_amdgcn_global_load_async_to_lds_b128((v4i*)g, (v4i*)l, 0, 0);
#else
  *(u16x8*)l = *(const u16x8*)g;
#endif
}
__device__ __forceinline__ void wait_async_copies() {
#if HAS_ASYNC_LDS
#if __has_builtin(__builtin_amdgcn_s_wait_asynccnt)
  __builtin_amdgcn_s_wait_asynccnt(0);
#else
  asm volatile("s_wait_asynccnt 0x0" ::: "memory");
#endif
#endif
}

union Frag { u16x16 u; v16bf bf; u16x8 h[2]; };

// ---------------- pre-pass: f32 -> bf16 activations ----------------
__global__ __launch_bounds__(256) void cvt_bf16_kernel(const float* __restrict__ src,
                                                       unsigned short* __restrict__ dst,
                                                       int nquads) {
  int q = blockIdx.x * 256 + threadIdx.x;
  if (q >= nquads) return;
  const float4 f = *(const float4*)(src + (size_t)q * 4);
  *(unsigned long long*)(dst + (size_t)q * 4) = pack4bf(f.x, f.y, f.z, f.w);
}

// ---------------- pre-pass: fold BN gain a[d]=g*rsqrt(v+eps) into bf16 weights ----------------
__global__ __launch_bounds__(256) void cvt_scale_w_kernel(const float* __restrict__ W,
                                                          const float* __restrict__ bnp,
                                                          unsigned short* __restrict__ dst,
                                                          int nquads) {
  int q = blockIdx.x * 256 + threadIdx.x;
  if (q >= nquads) return;
  int e = q * 4;
  int d = e % COUT;                      // COUT divisible by 4: quad never straddles rows
  const float4 f = *(const float4*)(W + e);
  float a0 = bnp[d + 0] * rsqrtf(bnp[3 * COUT + d + 0] + EPSV);
  float a1 = bnp[d + 1] * rsqrtf(bnp[3 * COUT + d + 1] + EPSV);
  float a2 = bnp[d + 2] * rsqrtf(bnp[3 * COUT + d + 2] + EPSV);
  float a3 = bnp[d + 3] * rsqrtf(bnp[3 * COUT + d + 3] + EPSV);
  *(unsigned long long*)(dst + e) = pack4bf(f.x * a0, f.y * a1, f.z * a2, f.w * a3);
}

// ---------------- fused sparse-conv + folded-BN (+residual pseudo-offset) + relu ----------------
// RES=false: out_bf  = bf16(relu(conv(src,Ws) + b1))
// RES=true : out_f32 = relu(conv(src,Ws) + b2 + resSrc@Wrs + br)
template <bool RES>
__global__ __launch_bounds__(256)
void sparse_conv_bn_wmma(const unsigned short* __restrict__ src,   // bf16 [n,96]
                         const long long* __restrict__ nbr,        // [27,n]
                         const unsigned short* __restrict__ Ws,    // bf16 scaled [27,96,96]
                         const float* __restrict__ bnp,
                         const unsigned short* __restrict__ Wrs,   // bf16 scaled [96,96]
                         const float* __restrict__ bnr,
                         const unsigned short* __restrict__ resSrc,
                         unsigned short* __restrict__ out_bf,
                         float* __restrict__ out_f32,
                         int n) {
  __shared__ __align__(16) unsigned short sA[2][M_TILE * SA_STRIDE]; // gathered rows
  __shared__ __align__(16) unsigned short sB[2][CIN * SB_STRIDE];    // weights
  __shared__ int   sIdx[2][M_TILE];
  __shared__ float sBias[COUT], sBiasR[COUT];

  const int tid  = threadIdx.x;
  const int row0 = blockIdx.x * M_TILE;
  const int passes = RES ? (KOFF + 1) : KOFF;

  if (tid < COUT) {
    float g = bnp[tid], b = bnp[COUT + tid], m = bnp[2 * COUT + tid], v = bnp[3 * COUT + tid];
    sBias[tid] = b - g * rsqrtf(v + EPSV) * m;
  }
  if (RES && tid >= 128 && tid < 128 + COUT) {
    int d = tid - 128;
    float g = bnr[d], b = bnr[COUT + d], m = bnr[2 * COUT + d], v = bnr[3 * COUT + d];
    sBiasR[d] = b - g * rsqrtf(v + EPSV) * m;
  }

  auto stage_idx = [&](int pass, int buf) {
    if (tid < M_TILE) {
      int row = row0 + tid;
      if (row >= n) row = n - 1;
      bool rp = RES && (pass == KOFF);
      sIdx[buf][tid] = rp ? row : (int)nbr[(long long)pass * n + row];
    }
  };
  auto issue_copies = [&](int pass, int buf) {
    const bool rp = RES && (pass == KOFF);
    const unsigned short* gs = rp ? resSrc : src;
    const unsigned short* Wk = rp ? Wrs : (Ws + (size_t)pass * CIN * COUT);
    const int* idx = sIdx[buf];
    unsigned short* bA = sA[buf];
    unsigned short* bB = sB[buf];
#pragma unroll 1
    for (int t = tid; t < M_TILE * 12; t += 256) {        // 128 rows x 12 b128 chunks
      int r = t / 12, p = t - r * 12;
      copy_chunk_b128(gs + (size_t)idx[r] * CIN + p * 8, bA + r * SA_STRIDE + p * 8);
    }
#pragma unroll 1
    for (int t = tid; t < CIN * 12; t += 256) {           // 96 rows x 12 b128 chunks
      int c = t / 12, p = t - c * 12;
      copy_chunk_b128(Wk + c * COUT + p * 8, bB + c * SB_STRIDE + p * 8);
    }
  };

  const int lane = tid & 31;
  const int wave = tid >> 5;
  const int half = lane >> 4;
  const int ml   = lane & 15;
  const int arow = wave * 16 + ml;

  v8f acc[6];
  const v8f zero = {0.f, 0.f, 0.f, 0.f, 0.f, 0.f, 0.f, 0.f};
#pragma unroll
  for (int i = 0; i < 6; ++i) acc[i] = zero;

  // prologue: stage pass 0
  stage_idx(0, 0);
  __syncthreads();
  issue_copies(0, 0);

  int cur = 0;
  for (int kk = 0; kk < passes; ++kk) {
    const int nxt = cur ^ 1;
    if (kk + 1 < passes) stage_idx(kk + 1, nxt);

    wait_async_copies();     // my DMA for buf[cur] landed
    __syncthreads();         // everyone's landed; sIdx[nxt] visible

    if (kk + 1 < passes) issue_copies(kk + 1, nxt);  // overlap next DMA with compute

    const unsigned short* bA = sA[cur];
    const unsigned short* bB = sB[cur];
#pragma unroll
    for (int ct = 0; ct < 3; ++ct) {
      Frag A;   // A 16x32 bf16: half=0 -> K {0..7,16..23}; half=1 -> K {8..15,24..31}
      A.h[0] = *(const u16x8*)(&bA[arow * SA_STRIDE + ct * 32 + 8 * half]);
      A.h[1] = *(const u16x8*)(&bA[arow * SA_STRIDE + ct * 32 + 16 + 8 * half]);
      const int bc = ct * 32 + 8 * half;  // B: lane carries column N; elements K {8h..8h+7, 16+8h..16+8h+7}
#pragma unroll
      for (int nt = 0; nt < 6; ++nt) {
        Frag B;
#pragma unroll
        for (int s = 0; s < 8; ++s) {
          B.u[s]     = bB[(bc + s) * SB_STRIDE + nt * 16 + ml];
          B.u[8 + s] = bB[(bc + 16 + s) * SB_STRIDE + nt * 16 + ml];
        }
        acc[nt] = __builtin_amdgcn_wmma_f32_16x16x32_bf16(
            false, A.bf, false, B.bf, (short)0, acc[nt], false, false);
      }
    }
    cur = nxt;
  }

  // epilogue: bias + relu; C/D layout: vgpr j -> row j+8*half, lane&15 -> col
  const int rbase = row0 + wave * 16 + 8 * half;
#pragma unroll
  for (int nt = 0; nt < 6; ++nt) {
    const int col = nt * 16 + ml;
    float bb = sBias[col];
    if (RES) bb += sBiasR[col];
#pragma unroll
    for (int j = 0; j < 8; ++j) {
      int r = rbase + j;
      if (r < n) {
        float v = fmaxf(acc[nt][j] + bb, 0.0f);
        if (RES) out_f32[(size_t)r * COUT + col] = v;
        else     out_bf[(size_t)r * COUT + col] = f2bf(v);
      }
    }
  }
}

extern "C" void kernel_launch(void* const* d_in, const int* in_sizes, int n_in,
                              void* d_out, int out_size, void* d_ws, size_t ws_size,
                              hipStream_t stream) {
  const float*     feats = (const float*)d_in[0];
  const long long* nbr   = (const long long*)d_in[1];
  const float*     W1    = (const float*)d_in[2];
  const float*     W2    = (const float*)d_in[3];
  const float*     Wr    = (const float*)d_in[4];
  const float*     bn1   = (const float*)d_in[5];
  const float*     bn2   = (const float*)d_in[6];
  const float*     bnr   = (const float*)d_in[7];
  float*           out   = (float*)d_out;

  const int    n     = in_sizes[0] / CIN;       // 65536
  const size_t nElem = (size_t)n * CIN;

  // workspace layout (u16 units, all segments 16B aligned)
  unsigned short* featsBF = (unsigned short*)d_ws;
  unsigned short* hBF     = featsBF + nElem;
  unsigned short* W1s     = hBF + nElem;
  unsigned short* W2s     = W1s + (size_t)KOFF * CIN * COUT;
  unsigned short* Wrs     = W2s + (size_t)KOFF * CIN * COUT;

  // pre-passes: convert activations, fold BN gains into bf16 weights (cheap, every call)
  {
    int q = (int)(nElem / 4);
    cvt_bf16_kernel<<<(q + 255) / 256, 256, 0, stream>>>(feats, featsBF, q);
    int qw = KOFF * CIN * COUT / 4;
    cvt_scale_w_kernel<<<(qw + 255) / 256, 256, 0, stream>>>(W1, bn1, W1s, qw);
    cvt_scale_w_kernel<<<(qw + 255) / 256, 256, 0, stream>>>(W2, bn2, W2s, qw);
    int qr = CIN * COUT / 4;
    cvt_scale_w_kernel<<<(qr + 255) / 256, 256, 0, stream>>>(Wr, bnr, Wrs, qr);
  }

  dim3 grid((n + M_TILE - 1) / M_TILE), block(256);
  // conv1 + bn1 + relu -> h (bf16, workspace)
  sparse_conv_bn_wmma<false><<<grid, block, 0, stream>>>(
      featsBF, nbr, W1s, bn1, nullptr, nullptr, nullptr, hBF, nullptr, n);
  // conv2 + bn2 + residual(featsBF@Wrs, bnr folded) + relu -> out (f32)
  sparse_conv_bn_wmma<true><<<grid, block, 0, stream>>>(
      hBF, nbr, W2s, bn2, Wrs, bnr, featsBF, nullptr, out, n);
}
